// EmbeddingBlock_52166672778209
// MI455X (gfx1250) — matrix operand order repaired and checked by reading, not compiled, not run on hardware
//
#include <hip/hip_runtime.h>
#include <hip/hip_bf16.h>

#define EMB 128
#define TILE 16
#define MT 2            // M-tiles (16 edges each) per block iteration => 32 edges
#define ITERS 32        // iterations per block => 1024 edges/block
#define ASTRIDE 136     // padded bf16 row stride (LDS bank decorrelation)
#define FSTRIDE 132     // padded f32 table row stride (LDS bank decorrelation)

typedef __attribute__((ext_vector_type(16))) __bf16        v16bf;
typedef __attribute__((ext_vector_type(8)))  float         v8f;
typedef __attribute__((ext_vector_type(8)))  unsigned int  v8u;

__device__ __forceinline__ unsigned short f2bf(float f) {
    unsigned int u = __float_as_uint(f);
    u += 0x7fffu + ((u >> 16) & 1u);            // round-to-nearest-even
    return (unsigned short)(u >> 16);
}

__device__ __forceinline__ float silu(float x) {
    return x / (1.0f + __expf(-x));
}

// ---------------------------------------------------------------------------
// P1: fold all linear concat blocks into tiny gather tables (214 x 128 f32):
//   rows [0,96)    embW0 = emb_table @ W[0:128]
//   rows [96,192)  embW1 = emb_table @ W[128:256]
//   rows [192,212) stW3  = step_table @ W[384:512]
//   rows [212,214) eqW4  = equiv_table @ W[512:640] + b   (bias folded here)
// ---------------------------------------------------------------------------
__global__ void precompute_tables(const float* __restrict__ emb,
                                  const float* __restrict__ stepT,
                                  const float* __restrict__ eqT,
                                  const float* __restrict__ W,
                                  const float* __restrict__ b,
                                  float* __restrict__ tbl) {
    __shared__ float row[EMB];
    const int r = blockIdx.x, t = threadIdx.x;
    const float* srcRow;
    int wOff;
    bool addB = false;
    if (r < 96)       { srcRow = emb   + (size_t)r * EMB;         wOff = 0;   }
    else if (r < 192) { srcRow = emb   + (size_t)(r - 96) * EMB;  wOff = 128; }
    else if (r < 212) { srcRow = stepT + (size_t)(r - 192) * EMB; wOff = 384; }
    else              { srcRow = eqT   + (size_t)(r - 212) * EMB; wOff = 512; addB = true; }
    row[t] = srcRow[t];
    __syncthreads();
    float acc = addB ? b[t] : 0.0f;
    for (int k = 0; k < EMB; ++k)
        acc += row[k] * W[(size_t)(wOff + k) * EMB + t];
    tbl[(size_t)r * EMB + t] = acc;
}

// P2: W2^T in bf16 (w2t[n*128+k] = bf16(W[(256+k)*128 + n])) -> WMMA B operand
__global__ void precompute_w2t(const float* __restrict__ W,
                               unsigned short* __restrict__ w2t) {
    int idx = blockIdx.x * 256 + threadIdx.x;   // 0..16383
    int n = idx >> 7, k = idx & 127;
    w2t[idx] = f2bf(W[(size_t)(256 + k) * EMB + n]);
}

// P3: bessel basis * DimeNet envelope (p = 6), streaming over edges
__global__ void radial_env_kernel(const float* __restrict__ d,
                                  float* __restrict__ outEnv, int E) {
    int t = blockIdx.x * 256 + threadIdx.x;
    if (t >= E) return;
    float x   = d[t] * 0.2f;                    // d / CUTOFF
    float inv = 1.0f / x;
    float x2 = x * x, x4 = x2 * x2, x5 = x4 * x;
    float env = inv - 28.0f * x5 + 48.0f * x5 * x - 21.0f * x5 * x2;
    const float PI = 3.14159265358979323846f;
    float* o = outEnv + (size_t)t * 6;
#pragma unroll
    for (int r = 0; r < 6; ++r) {
        float s = __sinf(x * (PI * (float)(r + 1)));
        o[r] = env * s * inv;                   // env(x) * sin(f*x)/x
    }
}

// ---------------------------------------------------------------------------
// Main edge kernel: per 32-edge iteration (2 WMMA M-tiles, B frags reused)
//   rbf_a = silu(rbf @ W_rbf + b_rbf)            (VALU, bf16 into LDS)
//   m     = silu(rbf_a @ W2 + 4 LDS table gathers) via v_wmma_f32_16x16x32_bf16
// All tables live in LDS (~161 KB => 2 blocks/WGP).
// ---------------------------------------------------------------------------
__global__ void __launch_bounds__(256, 2)
edge_msg_kernel(const int* __restrict__ Z, const int* __restrict__ stepA,
                const int* __restrict__ src, const int* __restrict__ dst,
                const int* __restrict__ equiv, const float* __restrict__ rbf,
                const float* __restrict__ wrbf, const float* __restrict__ brbf,
                const float* __restrict__ tbl, const unsigned short* __restrict__ w2t,
                float* __restrict__ out, int E, int niters) {
    extern __shared__ char smem[];
    float* embW0L = (float*)smem;                    // 96 rows  x FSTRIDE
    float* embW1L = embW0L + 96 * FSTRIDE;           // 96 rows
    float* stW3L  = embW1L + 96 * FSTRIDE;           // 20 rows
    float* eqW4L  = stW3L + 20 * FSTRIDE;            // 2 rows (bias folded)
    float* wrbfL  = eqW4L + 2 * FSTRIDE;             // 6*128
    float* brbfL  = wrbfL + 6 * EMB;                 // 128
    float* rbfT   = brbfL + EMB;                     // 32*6
    int*   idxT   = (int*)(rbfT + MT * TILE * 6);    // 32*4
    unsigned short* w2L = (unsigned short*)(idxT + MT * TILE * 4); // 128 x ASTRIDE
    unsigned short* aL  = w2L + EMB * ASTRIDE;                     // 32 x ASTRIDE

    const int tid = threadIdx.x;

    // ---- preload: 214 padded f32 table rows, W_rbf/b_rbf, W2^T bf16 (padded)
    {
        for (int i = tid; i < 214 * 32; i += 256) {      // row-padded f32 copy
            int row = i >> 5, c = i & 31;
            ((float4*)(embW0L + row * FSTRIDE))[c] = ((const float4*)(tbl + row * EMB))[c];
        }
        const float4* wr4 = (const float4*)wrbf;
        float4* wd4 = (float4*)wrbfL;
        for (int i = tid; i < (6 * EMB) / 4; i += 256) wd4[i] = wr4[i];
        if (tid < EMB / 4) ((float4*)brbfL)[tid] = ((const float4*)brbf)[tid];
        for (int i = tid; i < EMB * 32; i += 256) {      // 128 rows x 32 uint4
            int n = i >> 5, c = i & 31;
            ((uint4*)(w2L + n * ASTRIDE))[c] = ((const uint4*)(w2t + n * EMB))[c];
        }
    }
    __syncthreads();

    const int lane   = tid & 31;
    const int w      = tid >> 5;          // 8 waves -> 8 N-slabs of 16
    const int laneHi = (lane >> 4) & 1;
    const int lm     = lane & 15;
    const int ncol   = w * 16 + lm;

    int it0 = blockIdx.x * ITERS;
    int it1 = it0 + ITERS; if (it1 > niters) it1 = niters;

    for (int it = it0; it < it1; ++it) {
        const long base = (long)it * (MT * TILE);    // 32 edges / iteration

        // stage per-edge indices + rbf tile
        if (tid < MT * TILE) {
            long e = base + tid; if (e >= E) e = E - 1;
            int s = src[e];
            idxT[tid * 4 + 0] = Z[s];
            idxT[tid * 4 + 1] = Z[dst[e]];
            idxT[tid * 4 + 2] = stepA[s];
            idxT[tid * 4 + 3] = equiv[e];
        }
        if (tid < MT * TILE * 6) {
            int e = tid / 6, r = tid - e * 6;
            long eg = base + e; if (eg >= E) eg = E - 1;
            rbfT[tid] = rbf[eg * 6 + r];
        }
        __syncthreads();

        // rbf_a = silu(rbf @ W_rbf + b_rbf) -> bf16 A tiles in LDS (32 x 128)
        {
            int e = tid & 31, k0 = tid >> 5;     // 8 k-phases, 16 k per thread
            float r0 = rbfT[e * 6 + 0], r1 = rbfT[e * 6 + 1], r2 = rbfT[e * 6 + 2];
            float r3 = rbfT[e * 6 + 3], r4 = rbfT[e * 6 + 4], r5 = rbfT[e * 6 + 5];
#pragma unroll
            for (int j = 0; j < 16; ++j) {
                int k = k0 + j * 8;
                float a = brbfL[k]
                        + r0 * wrbfL[0 * EMB + k] + r1 * wrbfL[1 * EMB + k]
                        + r2 * wrbfL[2 * EMB + k] + r3 * wrbfL[3 * EMB + k]
                        + r4 * wrbfL[4 * EMB + k] + r5 * wrbfL[5 * EMB + k];
                aL[e * ASTRIDE + k] = f2bf(silu(a));
            }
        }
        __syncthreads();

        // [32 x 128] x [128 x 16]: 8x v_wmma_f32_16x16x32_bf16, B frags reused
        v8f acc[MT] = {};
#pragma unroll
        for (int ks = 0; ks < 4; ++ks) {
            v8u bu;
#pragma unroll
            for (int p = 0; p < 8; ++p) {
                int i  = p * 2;
                int Kb = ks * 32 + ((i >> 3) << 4) + laneHi * 8 + (i & 7); // B 32x16: K = (i/8)*16 + 8*laneHi + i%8
                bu[p] = *(const unsigned int*)(w2L + ncol * ASTRIDE + Kb);
            }
#pragma unroll
            for (int mt = 0; mt < MT; ++mt) {
                v8u au;
#pragma unroll
                for (int p = 0; p < 8; ++p) {
                    int i  = p * 2;
                    // A 16x32: K = (i/8)*16 + 8*laneHi + i%8
                    int Ka = ks * 32 + ((i >> 3) << 4) + laneHi * 8 + (i & 7);
                    au[p] = *(const unsigned int*)(aL + (mt * TILE + lm) * ASTRIDE + Ka);
                }
                acc[mt] = __builtin_amdgcn_wmma_f32_16x16x32_bf16(
                    false, __builtin_bit_cast(v16bf, au),
                    false, __builtin_bit_cast(v16bf, bu),
                    (short)0, acc[mt], false, false);
            }
        }

        // epilogue: + gathered linear terms (bias inside eqW4L), silu, store
        const bool full = (base + MT * TILE) <= (long)E;
#pragma unroll
        for (int mt = 0; mt < MT; ++mt) {
#pragma unroll
            for (int r = 0; r < 8; ++r) {
                int e = mt * TILE + r + laneHi * 8;   // C/D layout: M=r+8*laneHi
                long eg = base + e;
                int zs = idxT[e * 4 + 0], zd = idxT[e * 4 + 1];
                int ss = idxT[e * 4 + 2], qq = idxT[e * 4 + 3];
                float v = acc[mt][r]
                        + embW0L[zs * FSTRIDE + ncol] + embW1L[zd * FSTRIDE + ncol]
                        + stW3L[ss * FSTRIDE + ncol]  + eqW4L[qq * FSTRIDE + ncol];
                v = silu(v);
                if (full || eg < E) out[eg * (long)EMB + ncol] = v;
            }
        }
        __syncthreads();
    }
}

extern "C" void kernel_launch(void* const* d_in, const int* in_sizes, int n_in,
                              void* d_out, int out_size, void* d_ws, size_t ws_size,
                              hipStream_t stream) {
    const int*   Z     = (const int*)d_in[0];
    const int*   stepA = (const int*)d_in[1];
    const int*   src   = (const int*)d_in[2];
    const int*   dst   = (const int*)d_in[3];
    const int*   equiv = (const int*)d_in[4];
    const float* rbf   = (const float*)d_in[5];
    const float* dvec  = (const float*)d_in[6];
    const float* embT  = (const float*)d_in[7];
    const float* stepT = (const float*)d_in[8];
    const float* eqT   = (const float*)d_in[9];
    const float* wrbf  = (const float*)d_in[10];
    const float* brbf  = (const float*)d_in[11];
    const float* W     = (const float*)d_in[12];
    const float* b     = (const float*)d_in[13];
    const int E = in_sizes[2];
    float* out = (float*)d_out;

    float* tbl = (float*)d_ws;                                        // 214*128 f32
    unsigned short* w2t =
        (unsigned short*)((char*)d_ws + (size_t)214 * EMB * sizeof(float)); // 128*128 bf16

    precompute_tables<<<214, 128, 0, stream>>>(embT, stepT, eqT, W, b, tbl);
    precompute_w2t<<<64, 256, 0, stream>>>(W, w2t);
    radial_env_kernel<<<(E + 255) / 256, 256, 0, stream>>>(dvec, out + (size_t)E * EMB, E);

    const int niters  = (E + MT * TILE - 1) / (MT * TILE);
    const int nblocks = (niters + ITERS - 1) / ITERS;
    constexpr size_t SMEM =
        (size_t)(214 * FSTRIDE + 6 * EMB + EMB + MT * TILE * 6) * sizeof(float)
        + (size_t)(MT * TILE * 4) * sizeof(int)
        + (size_t)(EMB * ASTRIDE + MT * TILE * ASTRIDE) * sizeof(unsigned short);
    (void)hipFuncSetAttribute((const void*)edge_msg_kernel,
                              hipFuncAttributeMaxDynamicSharedMemorySize, (int)SMEM);
    edge_msg_kernel<<<nblocks, 256, SMEM, stream>>>(
        Z, stepA, src, dst, equiv, rbf, wrbf, brbf, tbl, w2t, out, E, niters);
}
